// GATv2ConvNet_22935125360682
// MI455X (gfx1250) — compile-verified, matrix operand order checked by reading, not run on hardware
//
#include <hip/hip_runtime.h>
#include <hip/hip_bf16.h>
#include <cstdint>
#include <cstddef>

typedef __attribute__((ext_vector_type(16))) _Float16 v16h;
typedef __attribute__((ext_vector_type(8)))  float    v8f;

#define NEG_SLOPE 0.2f

// ---- monotone float<->uint encoding for atomic segment-max (handles signs) ----
__device__ __forceinline__ unsigned fenc(float f) {
  unsigned u = __float_as_uint(f);
  return (u & 0x80000000u) ? ~u : (u | 0x80000000u);
}
__device__ __forceinline__ float fdec(unsigned e) {
  return (e & 0x80000000u) ? __uint_as_float(e & 0x7fffffffu) : __uint_as_float(~e);
}
#define ENC_NEG_INF 0x007FFFFFu   // fenc(-inf)

// ---------------------------------------------------------------------------
// Utility fills
// ---------------------------------------------------------------------------
__global__ void fill_f32(float* __restrict__ p, float v, long n) {
  long i = (long)blockIdx.x * blockDim.x + threadIdx.x;
  if (i < n) p[i] = v;
}
__global__ void fill_u32(unsigned* __restrict__ p, unsigned v, long n) {
  long i = (long)blockIdx.x * blockDim.x + threadIdx.x;
  if (i < n) p[i] = v;
}
template <int F>
__global__ void bias_broadcast(float* __restrict__ out, const float* __restrict__ bias,
                               long n) {
  long i = (long)blockIdx.x * blockDim.x + threadIdx.x;
  if (i < n) out[i] = bias[(int)(i % F)];
}

// ---------------------------------------------------------------------------
// Edge list with self-loops appended (PyG add_self_loops=True)
// ---------------------------------------------------------------------------
__global__ void build_edges(const int* __restrict__ edge_index,
                            int* __restrict__ src, int* __restrict__ dst,
                            int E, int Etot) {
  int i = blockIdx.x * blockDim.x + threadIdx.x;
  if (i >= Etot) return;
  if (i < E) {
    src[i] = edge_index[i];
    dst[i] = edge_index[E + i];
  } else {
    src[i] = i - E;
    dst[i] = i - E;
  }
}

// ---------------------------------------------------------------------------
// Embedding gather into K=32 zero-padded rows
// ---------------------------------------------------------------------------
__global__ void gather_emb32(const float* __restrict__ emb, const int* __restrict__ ids,
                             float* __restrict__ x, long n32) {
  long i = (long)blockIdx.x * blockDim.x + threadIdx.x;
  if (i >= n32) return;
  const int j = (int)(i & 31);
  if (j < 16) {
    int node = ids[i >> 5];
    x[i] = emb[(long)node * 16 + j];
  } else {
    x[i] = 0.f;
  }
}

// ---------------------------------------------------------------------------
// Pack weights B[Kreal, NOUT] (fp32) into f16 WMMA B-fragment order, zero pad
// to Kpad. Flat layout: Bp[ ((t*NOUT + n)*2 + hi)*16 + j ], where
// k = t*32 + (j<8 ? hi*8+j : 16+hi*8+(j-8))   (ISA 7.12.2 16-bit layout).
// ---------------------------------------------------------------------------
__global__ void pack_B(const float* __restrict__ B, _Float16* __restrict__ Bp,
                       int Kreal, int Kpad, int Nout) {
  long idx = (long)blockIdx.x * blockDim.x + threadIdx.x;
  const long total = (long)Kpad * Nout;
  if (idx >= total) return;
  const int j  = (int)(idx & 15);
  const int hi = (int)((idx >> 4) & 1);
  const long rest = idx >> 5;
  const int n = (int)(rest % Nout);
  const int t = (int)(rest / Nout);
  const int k = t * 32 + ((j < 8) ? (hi * 8 + j) : (16 + hi * 8 + (j - 8)));
  Bp[idx] = (k < Kreal) ? (_Float16)B[(size_t)k * Nout + n] : (_Float16)0.f;
}

// ---------------------------------------------------------------------------
// WMMA GEMM + bias:  C[Nrows, NOUT] = A[Nrows, K](fp32) * Bp(f16 packed) + bias
// 4 waves/block; each wave computes one 16x32 output tile: A fragment is built
// once per k-step (4 x b128 + pk-cvt) and reused for two WMMAs.
// ---------------------------------------------------------------------------
template <int K, int NOUT>
__global__ void wmma_gemm_bias(const float* __restrict__ A,
                               const _Float16* __restrict__ Bp,
                               const float* __restrict__ bias,
                               float* __restrict__ C, int Nrows) {
  const int lane = threadIdx.x & 31;
  const int wave = threadIdx.x >> 5;
  const int lo   = lane & 15;
  const int hi   = lane >> 4;                 // 0 or 1
  const int row0 = (blockIdx.x * 4 + wave) * 16;
  const int col0 = blockIdx.y * 32;           // two 16-wide tiles per wave
  int arow = row0 + lo;
  if (arow >= Nrows) arow = Nrows - 1;        // clamp: loads stay in-bounds

  const float* __restrict__ Arow = A + (size_t)arow * K;
  const _Float16* __restrict__ Bl0 = Bp + ((size_t)(col0 + lo) * 2 + hi) * 16;
  const _Float16* __restrict__ Bl1 = Bp + ((size_t)(col0 + 16 + lo) * 2 + hi) * 16;

  v8f acc0 = {}, acc1 = {};
#pragma unroll
  for (int t = 0; t < K / 32; ++t) {
    const float4* pa0 = (const float4*)(Arow + t * 32 + hi * 8);
    const float4* pa1 = (const float4*)(Arow + t * 32 + 16 + hi * 8);
    const float4 x0 = pa0[0], x1 = pa0[1];
    const float4 x2 = pa1[0], x3 = pa1[1];
    v16h a;
    a[0]  = (_Float16)x0.x; a[1]  = (_Float16)x0.y;
    a[2]  = (_Float16)x0.z; a[3]  = (_Float16)x0.w;
    a[4]  = (_Float16)x1.x; a[5]  = (_Float16)x1.y;
    a[6]  = (_Float16)x1.z; a[7]  = (_Float16)x1.w;
    a[8]  = (_Float16)x2.x; a[9]  = (_Float16)x2.y;
    a[10] = (_Float16)x2.z; a[11] = (_Float16)x2.w;
    a[12] = (_Float16)x3.x; a[13] = (_Float16)x3.y;
    a[14] = (_Float16)x3.z; a[15] = (_Float16)x3.w;

    const v16h b0 = *(const v16h*)(Bl0 + (size_t)t * NOUT * 32);
    const v16h b1 = *(const v16h*)(Bl1 + (size_t)t * NOUT * 32);

    // 8 args: (neg_a, A, neg_b, B, c_mod, C, reuse_a, reuse_b)
    acc0 = __builtin_amdgcn_wmma_f32_16x16x32_f16(false, a, false, b0, (short)0, acc0,
                                                  false, false);
    acc1 = __builtin_amdgcn_wmma_f32_16x16x32_f16(false, a, false, b1, (short)0, acc1,
                                                  false, false);
  }

  const float bv0 = bias[col0 + lo];
  const float bv1 = bias[col0 + 16 + lo];
  float* __restrict__ Cl = C + (size_t)(row0 + hi * 8) * NOUT + col0 + lo;
  if (row0 + 16 <= Nrows) {
#pragma unroll
    for (int v = 0; v < 8; ++v) {
      Cl[(size_t)v * NOUT]      = acc0[v] + bv0;
      Cl[(size_t)v * NOUT + 16] = acc1[v] + bv1;
    }
  } else {
#pragma unroll
    for (int v = 0; v < 8; ++v) {
      if (row0 + hi * 8 + v < Nrows) {
        Cl[(size_t)v * NOUT]      = acc0[v] + bv0;
        Cl[(size_t)v * NOUT + 16] = acc1[v] + bv1;
      }
    }
  }
}

// ---------------------------------------------------------------------------
// Pass 1: per (edge, head) attention logit + segment max over dst (float4 path)
// ---------------------------------------------------------------------------
template <int H, int C>
__global__ void edge_logits_max(const int* __restrict__ src, const int* __restrict__ dst,
                                const float* __restrict__ xl, const float* __restrict__ xr,
                                const float* __restrict__ att,
                                float* __restrict__ logit, unsigned* __restrict__ mx,
                                long Etot) {
  long idx = (long)blockIdx.x * blockDim.x + threadIdx.x;
  if (idx >= Etot * H) return;
  const long e = idx / H;
  const int  h = (int)(idx % H);
  const int  s = src[e];
  const int  d = dst[e];
  const float4* pl = (const float4*)(xl + (size_t)s * (H * C) + h * C);
  const float4* pr = (const float4*)(xr + (size_t)d * (H * C) + h * C);
  const float4* pa = (const float4*)(att + h * C);
  float acc = 0.f;
#pragma unroll
  for (int c = 0; c < C / 4; ++c) {
    const float4 l = pl[c], r = pr[c], a4 = pa[c];
    float v;
    v = l.x + r.x; v = (v > 0.f) ? v : NEG_SLOPE * v; acc += a4.x * v;
    v = l.y + r.y; v = (v > 0.f) ? v : NEG_SLOPE * v; acc += a4.y * v;
    v = l.z + r.z; v = (v > 0.f) ? v : NEG_SLOPE * v; acc += a4.z * v;
    v = l.w + r.w; v = (v > 0.f) ? v : NEG_SLOPE * v; acc += a4.w * v;
  }
  logit[idx] = acc;
  atomicMax(&mx[(size_t)d * H + h], fenc(acc));
}

// ---------------------------------------------------------------------------
// Pass 2: p = exp(logit - max[dst]); denom[dst] += p  (p overwrites logit)
// ---------------------------------------------------------------------------
template <int H>
__global__ void edge_exp_denom(const int* __restrict__ dst,
                               float* __restrict__ logit,
                               const unsigned* __restrict__ mx,
                               float* __restrict__ denom,
                               long Etot) {
  long idx = (long)blockIdx.x * blockDim.x + threadIdx.x;
  if (idx >= Etot * H) return;
  const long e = idx / H;
  const int  h = (int)(idx % H);
  const int  d = dst[e];
  const float m = fdec(mx[(size_t)d * H + h]);
  const float p = __expf(logit[idx] - m);
  logit[idx] = p;
  atomicAdd(&denom[(size_t)d * H + h], p);
}

// ---------------------------------------------------------------------------
// Pass 3: out[dst, f] += xl[src, f] * (p / denom[dst, head(f)])
// ---------------------------------------------------------------------------
template <int H, int C>
__global__ void edge_aggregate(const int* __restrict__ src, const int* __restrict__ dst,
                               const float* __restrict__ xl,
                               const float* __restrict__ p,
                               const float* __restrict__ denom,
                               float* __restrict__ out,
                               long Etot) {
  constexpr int HC = H * C;
  long idx = (long)blockIdx.x * blockDim.x + threadIdx.x;
  if (idx >= Etot * HC) return;
  const long e = idx / HC;
  const int  f = (int)(idx % HC);
  const int  h = f / C;
  const int  s = src[e];
  const int  d = dst[e];
  const float alpha = p[e * H + h] / denom[(size_t)d * H + h];
  atomicAdd(&out[(size_t)d * HC + f], xl[(size_t)s * HC + f] * alpha);
}

// ---------------------------------------------------------------------------
// Global mean pool (accumulate phase)
// ---------------------------------------------------------------------------
__global__ void pool_accum(const float* __restrict__ x, const int* __restrict__ batch,
                           float* __restrict__ gsum, float* __restrict__ gcnt, long n64) {
  long idx = (long)blockIdx.x * blockDim.x + threadIdx.x;
  if (idx >= n64) return;
  const long i = idx >> 6;
  const int  f = (int)(idx & 63);
  const int  b = batch[i];
  atomicAdd(&gsum[(size_t)b * 64 + f], x[idx]);
  if (f == 0) atomicAdd(&gcnt[b], 1.f);
}

// ---------------------------------------------------------------------------
// Classifier head
// ---------------------------------------------------------------------------
__global__ void mlp_head(const float* __restrict__ gsum, const float* __restrict__ gcnt,
                         const float* __restrict__ demo,
                         const float* __restrict__ Wc1, const float* __restrict__ bc1,
                         const float* __restrict__ Wc2, const float* __restrict__ bc2,
                         float* __restrict__ out, int G) {
  int g = blockIdx.x * blockDim.x + threadIdx.x;
  if (g >= G) return;
  float in[69];
  const float cnt = fmaxf(gcnt[g], 1.f);
  for (int f = 0; f < 64; ++f) in[f] = gsum[(size_t)g * 64 + f] / cnt;
  for (int f = 0; f < 5; ++f)  in[64 + f] = demo[(size_t)g * 5 + f];
  float h[32];
  for (int k = 0; k < 32; ++k) {
    float s = bc1[k];
    for (int j = 0; j < 69; ++j) s += in[j] * Wc1[(size_t)j * 32 + k];
    h[k] = (s > 0.f) ? s : 0.f;
  }
  for (int o = 0; o < 2; ++o) {
    float s = bc2[o];
    for (int k = 0; k < 32; ++k) s += h[k] * Wc2[(size_t)k * 2 + o];
    out[(size_t)g * 2 + o] = s;
  }
}

// ---------------------------------------------------------------------------
// host driver
// ---------------------------------------------------------------------------
static inline long cdivl(long a, long b) { return (a + b - 1) / b; }

template <int H, int C>
static void run_attention(const float* attW, const float* boW, float* outF,
                          const int* srcA, const int* dstA,
                          const float* xlB, const float* xrB,
                          float* logit, unsigned* mx, float* denom,
                          int N, int Etot, hipStream_t stream) {
  constexpr int HC = H * C;
  const int T = 256;
  fill_u32<<<cdivl((long)N * H, T), T, 0, stream>>>(mx, ENC_NEG_INF, (long)N * H);
  fill_f32<<<cdivl((long)N * H, T), T, 0, stream>>>(denom, 0.f, (long)N * H);
  edge_logits_max<H, C><<<cdivl((long)Etot * H, T), T, 0, stream>>>(
      srcA, dstA, xlB, xrB, attW, logit, mx, (long)Etot);
  edge_exp_denom<H><<<cdivl((long)Etot * H, T), T, 0, stream>>>(
      dstA, logit, mx, denom, (long)Etot);
  bias_broadcast<HC><<<cdivl((long)N * HC, T), T, 0, stream>>>(outF, boW, (long)N * HC);
  edge_aggregate<H, C><<<cdivl((long)Etot * HC, T), T, 0, stream>>>(
      srcA, dstA, xlB, logit, denom, outF, (long)Etot);
}

extern "C" void kernel_launch(void* const* d_in, const int* in_sizes, int n_in,
                              void* d_out, int out_size, void* d_ws, size_t ws_size,
                              hipStream_t stream) {
  (void)n_in; (void)out_size; (void)ws_size;

  const float* emb  = (const float*)d_in[0];
  const float* Wl[3] = {(const float*)d_in[1], (const float*)d_in[7],  (const float*)d_in[13]};
  const float* bl[3] = {(const float*)d_in[2], (const float*)d_in[8],  (const float*)d_in[14]};
  const float* Wr[3] = {(const float*)d_in[3], (const float*)d_in[9],  (const float*)d_in[15]};
  const float* br[3] = {(const float*)d_in[4], (const float*)d_in[10], (const float*)d_in[16]};
  const float* att[3]= {(const float*)d_in[5], (const float*)d_in[11], (const float*)d_in[17]};
  const float* bo[3] = {(const float*)d_in[6], (const float*)d_in[12], (const float*)d_in[18]};
  const float* Wc1 = (const float*)d_in[19];
  const float* bc1 = (const float*)d_in[20];
  const float* Wc2 = (const float*)d_in[21];
  const float* bc2 = (const float*)d_in[22];
  const float* demo = (const float*)d_in[23];
  const int* node_ids   = (const int*)d_in[24];
  const int* edge_index = (const int*)d_in[25];
  const int* batch      = (const int*)d_in[26];
  float* out = (float*)d_out;

  const int N = in_sizes[24];
  const int E = in_sizes[25] / 2;
  const int G = in_sizes[23] / 5;
  const int Etot = E + N;

  // ---- workspace carve ----
  size_t off = 0;
  auto walloc = [&](size_t bytes) -> void* {
    void* p = (char*)d_ws + off;
    off += (bytes + 255) & ~(size_t)255;
    return p;
  };
  int*       srcA  = (int*)walloc((size_t)Etot * 4);
  int*       dstA  = (int*)walloc((size_t)Etot * 4);
  float*     featA = (float*)walloc((size_t)N * 192 * 4);
  float*     featB = (float*)walloc((size_t)N * 192 * 4);
  float*     xlB   = (float*)walloc((size_t)N * 192 * 4);
  float*     xrB   = (float*)walloc((size_t)N * 192 * 4);
  float*     logit = (float*)walloc((size_t)Etot * 3 * 4);
  unsigned*  mx    = (unsigned*)walloc((size_t)N * 3 * 4);
  float*     denom = (float*)walloc((size_t)N * 3 * 4);
  float*     gsum  = (float*)walloc((size_t)G * 64 * 4);
  float*     gcnt  = (float*)walloc((size_t)G * 4);
  _Float16*  BpL   = (_Float16*)walloc((size_t)192 * 192 * 2);
  _Float16*  BpR   = (_Float16*)walloc((size_t)192 * 192 * 2);

  const int T = 256;
  const unsigned gx = (unsigned)cdivl(cdivl(N, 16), 4);

  // edges with self loops; padded embedding gather (stride 32)
  build_edges<<<cdivl(Etot, T), T, 0, stream>>>(edge_index, srcA, dstA, E, Etot);
  gather_emb32<<<cdivl((long)N * 32, T), T, 0, stream>>>(emb, node_ids, featA, (long)N * 32);

  // ---- Layer 0: K=16 (padded 32), HC=96, H=3, C=32 ----
  pack_B<<<cdivl(32L * 96, T), T, 0, stream>>>(Wl[0], BpL, 16, 32, 96);
  pack_B<<<cdivl(32L * 96, T), T, 0, stream>>>(Wr[0], BpR, 16, 32, 96);
  {
    dim3 gg(gx, 96 / 32);
    wmma_gemm_bias<32, 96><<<gg, 128, 0, stream>>>(featA, BpL, bl[0], xlB, N);
    wmma_gemm_bias<32, 96><<<gg, 128, 0, stream>>>(featA, BpR, br[0], xrB, N);
  }
  run_attention<3, 32>(att[0], bo[0], featB, srcA, dstA, xlB, xrB,
                       logit, mx, denom, N, Etot, stream);

  // ---- Layer 1: K=96, HC=192, H=2, C=96 ----
  pack_B<<<cdivl(96L * 192, T), T, 0, stream>>>(Wl[1], BpL, 96, 96, 192);
  pack_B<<<cdivl(96L * 192, T), T, 0, stream>>>(Wr[1], BpR, 96, 96, 192);
  {
    dim3 gg(gx, 192 / 32);
    wmma_gemm_bias<96, 192><<<gg, 128, 0, stream>>>(featB, BpL, bl[1], xlB, N);
    wmma_gemm_bias<96, 192><<<gg, 128, 0, stream>>>(featB, BpR, br[1], xrB, N);
  }
  run_attention<2, 96>(att[1], bo[1], featA, srcA, dstA, xlB, xrB,
                       logit, mx, denom, N, Etot, stream);

  // ---- Layer 2: K=192, HC=64, H=1, C=64 ----
  pack_B<<<cdivl(192L * 64, T), T, 0, stream>>>(Wl[2], BpL, 192, 192, 64);
  pack_B<<<cdivl(192L * 64, T), T, 0, stream>>>(Wr[2], BpR, 192, 192, 64);
  {
    dim3 gg(gx, 64 / 32);
    wmma_gemm_bias<192, 64><<<gg, 128, 0, stream>>>(featA, BpL, bl[2], xlB, N);
    wmma_gemm_bias<192, 64><<<gg, 128, 0, stream>>>(featA, BpR, br[2], xrB, N);
  }
  run_attention<1, 64>(att[2], bo[2], featB, srcA, dstA, xlB, xrB,
                       logit, mx, denom, N, Etot, stream);

  // ---- global mean pool + classifier head ----
  fill_f32<<<cdivl((long)G * 64, T), T, 0, stream>>>(gsum, 0.f, (long)G * 64);
  fill_f32<<<cdivl((long)G, T), T, 0, stream>>>(gcnt, 0.f, (long)G);
  pool_accum<<<cdivl((long)N * 64, T), T, 0, stream>>>(featB, batch, gsum, gcnt, (long)N * 64);
  mlp_head<<<cdivl((long)G, 128), 128, 0, stream>>>(gsum, gcnt, demo, Wc1, bc1, Wc2, bc2, out, G);
}
